// SMMGCL_3221225472423
// MI455X (gfx1250) — matrix operand order repaired and checked by reading, not compiled, not run on hardware
//
#include <hip/hip_runtime.h>

// ---------------------------------------------------------------------------
// Types for CDNA5 WMMA
// ---------------------------------------------------------------------------
typedef __attribute__((ext_vector_type(16))) __bf16 v16bf;
typedef __attribute__((ext_vector_type(8)))  float  v8f;
typedef unsigned short us;

#define DEVINL __device__ __forceinline__

DEVINL us f2bf(float x) {
    unsigned u = __float_as_uint(x);
    unsigned r = u + 0x7FFFu + ((u >> 16) & 1u);   // round-to-nearest-even
    return (us)(r >> 16);
}

union FragU {            // build a v16bf operand from two 16B LDS reads
    uint4  q[2];
    v16bf  v;
};

// GEMM tiling
#define BM 128
#define BK 32
#define LDSW 40          // bf16 elems per LDS row (32 data + 8 pad -> 20-dword stride)

// ---------------------------------------------------------------------------
// Tile staging: global bf16 (row-major, K contiguous) -> LDS, zero-padded edges.
// `full` (block-uniform) selects the unguarded vector-copy fast path.
// ---------------------------------------------------------------------------
DEVINL void stage_tile(us* lds, const us* __restrict__ G,
                       int ld, int row0, int nrowsTile, int Rows, int k0, int K,
                       int tid, bool full) {
    const int chunks = nrowsTile * 4;                 // 4 x 16B chunks per row
    if (full) {
        for (int c = tid; c < chunks; c += 256) {
            const int r  = c >> 2;
            const int cc = c & 3;
            const us* p = G + (size_t)(row0 + r) * ld + (k0 + cc * 8);
            if (cc == 0 && k0 + BK < K) __builtin_prefetch(p + BK, 0, 1);
            *(uint4*)(lds + r * LDSW + cc * 8) = *(const uint4*)p;
        }
    } else {
        for (int c = tid; c < chunks; c += 256) {
            const int r  = c >> 2;
            const int cc = c & 3;
            const int gr = row0 + r;
            const int gk = k0 + cc * 8;
            uint4 val;
            if (gr < Rows && (gk + 8) <= K) {
                val = *(const uint4*)(G + (size_t)gr * ld + gk);
            } else {
                union { uint4 v4; us u[8]; } tmp;
                for (int i = 0; i < 8; ++i) {
                    const int k = gk + i;
                    tmp.u[i] = (gr < Rows && k < K) ? G[(size_t)gr * ld + k]
                                                    : (us)0;
                }
                val = tmp.v4;
            }
            *(uint4*)(lds + r * LDSW + cc * 8) = val;
        }
    }
}

// ---------------------------------------------------------------------------
// C[M,N] = act( A[M,K] * B[N,K]^T (+ C_prev) (+ bias) )
// 8 waves: 4 (m) x 2 (n); each wave computes 32 x (16*NT) via 2 x NT WMMAs.
// Epilogue can store f32 C, bf16 C, and/or bf16 C^T.
// ---------------------------------------------------------------------------
template<int NT, int ACT, bool ACCUM, bool BIAS, bool SF32, bool SBF, bool SBFT>
__global__ __launch_bounds__(256) void gemm_nt(
    const us* __restrict__ A, int lda,
    const us* __restrict__ B, int ldb,
    float* __restrict__ Cf, int ldc,
    us* __restrict__ Cb, int ldcb,
    us* __restrict__ Ct, int ldct,
    const float* __restrict__ bias,
    int M, int N, int K)
{
    constexpr int BNc = 32 * NT;                 // block N extent (2 waves in n)
    __shared__ us sA[2][BM * LDSW];
    __shared__ us sB[2][BNc * LDSW];

    const int tid  = threadIdx.x;
    const int lane = tid & 31;
    const int wid  = tid >> 5;
    const int wm   = wid >> 1;       // 0..3  (32 rows each)
    const int wn   = wid & 1;        // 0..1  (16*NT cols each)
    const int lr   = lane & 15;
    const int lh   = lane >> 4;

    const int m0 = blockIdx.y * BM;
    const int n0 = blockIdx.x * BNc;

    const bool fullAr = (m0 + BM)  <= M;
    const bool fullBr = (n0 + BNc) <= N;

    v8f acc[2][NT];
    for (int i = 0; i < 2; ++i)
        for (int j = 0; j < NT; ++j)
            for (int e = 0; e < 8; ++e) acc[i][j][e] = 0.0f;

    const int KT = (K + BK - 1) / BK;

    stage_tile(sA[0], A, lda, m0, BM,  M, 0, K, tid, fullAr && (BK <= K));
    stage_tile(sB[0], B, ldb, n0, BNc, N, 0, K, tid, fullBr && (BK <= K));
    __syncthreads();

    for (int kt = 0; kt < KT; ++kt) {
        const int cur = kt & 1, nxt = cur ^ 1;
        if (kt + 1 < KT) {
            const int k0s = (kt + 1) * BK;
            const bool kfull = (k0s + BK) <= K;
            stage_tile(sA[nxt], A, lda, m0, BM,  M, k0s, K, tid, fullAr && kfull);
            stage_tile(sB[nxt], B, ldb, n0, BNc, N, k0s, K, tid, fullBr && kfull);
        }
        FragU a[2], b[NT];
#pragma unroll
        for (int t = 0; t < 2; ++t) {
            // A frag (16x32 bf16): lane row = lr, K runs [lh*8,+8) and [16+lh*8,+8)
            const us* pa = sA[cur] + (wm * 32 + t * 16 + lr) * LDSW;
            a[t].q[0] = *(const uint4*)(pa + lh * 8);
            a[t].q[1] = *(const uint4*)(pa + 16 + lh * 8);
        }
#pragma unroll
        for (int j = 0; j < NT; ++j) {
            // B frag (32x16 bf16): lane col = lr, K runs [lh*8,+8) and [16+lh*8,+8)
            const us* pb = sB[cur] + (wn * 16 * NT + j * 16 + lr) * LDSW;
            b[j].q[0] = *(const uint4*)(pb + lh * 8);
            b[j].q[1] = *(const uint4*)(pb + 16 + lh * 8);
        }
#pragma unroll
        for (int i = 0; i < 2; ++i)
#pragma unroll
            for (int j = 0; j < NT; ++j)
                acc[i][j] = __builtin_amdgcn_wmma_f32_16x16x32_bf16(
                    false, a[i].v, false, b[j].v, (short)0, acc[i][j],
                    false, false);
        __syncthreads();
    }

    // Epilogue: C/D layout -> lane lr is column, vgpr r is row (lh*8 + r)
#pragma unroll
    for (int i = 0; i < 2; ++i) {
#pragma unroll
        for (int j = 0; j < NT; ++j) {
            const int mb = m0 + wm * 32 + i * 16 + lh * 8;
            const int nn = n0 + wn * 16 * NT + j * 16 + lr;
            if (nn >= N) continue;
            float bv = 0.0f;
            if (BIAS) bv = bias[nn];
#pragma unroll
            for (int r = 0; r < 8; ++r) {
                const int mm = mb + r;
                if (mm >= M) continue;
                float v = acc[i][j][r];
                if (ACCUM) v += Cf[(size_t)mm * ldc + nn];
                if (BIAS)  v += bv;
                if (ACT == 1) v = v > 0.0f ? v : 0.0f;
                if (ACT == 2) v = 1.0f / (1.0f + __expf(-v));
                if (SF32) Cf[(size_t)mm * ldc + nn]  = v;
                if (SBF)  Cb[(size_t)mm * ldcb + nn] = f2bf(v);
                if (SBFT) Ct[(size_t)nn * ldct + mm] = f2bf(v);
            }
        }
    }
}

// ---------------------------------------------------------------------------
// Elementwise converts
// ---------------------------------------------------------------------------
__global__ __launch_bounds__(256) void k_cvt_bf16(const float* __restrict__ in,
                                                  us* __restrict__ out, long n) {
    long i = ((long)blockIdx.x * 256 + threadIdx.x) * 4;
    if (i + 4 <= n) {
        float4 v = *(const float4*)(in + i);
        union { uint2 v2; us u[4]; } o;
        o.u[0] = f2bf(v.x); o.u[1] = f2bf(v.y);
        o.u[2] = f2bf(v.z); o.u[3] = f2bf(v.w);
        *(uint2*)(out + i) = o.v2;
    } else {
        for (long j = i; j < n; ++j) out[j] = f2bf(in[j]);
    }
}

__global__ __launch_bounds__(256) void k_cvtT_bf16(const float* __restrict__ in,
                                                   us* __restrict__ out,
                                                   int R, int C) {
    long i = (long)blockIdx.x * 256 + threadIdx.x;
    if (i < (long)R * C) {
        int r = (int)(i / C), c = (int)(i % C);
        out[(long)c * R + r] = f2bf(in[i]);
    }
}

// ---------------------------------------------------------------------------
// Attention fusion: per node, w_v = relu(x_v W1 + b1) W2; beta = softmax(w);
// out = beta0*x0 + beta1*x1.   One wave32 per node, 8 nodes per block.
// ---------------------------------------------------------------------------
__global__ __launch_bounds__(256) void attention_kernel(
    const float* __restrict__ X0, const float* __restrict__ X1,
    const float* __restrict__ W1, const float* __restrict__ b1,
    const float* __restrict__ W2,
    float* __restrict__ OutF, us* __restrict__ OutB, int Nn)
{
    __shared__ float sW1[64 * 64];
    __shared__ float sW2[64];
    __shared__ float sb1[64];
    for (int i = threadIdx.x; i < 64 * 64; i += 256) sW1[i] = W1[i];
    if (threadIdx.x < 64) { sW2[threadIdx.x] = W2[threadIdx.x];
                            sb1[threadIdx.x] = b1[threadIdx.x]; }
    __syncthreads();

    const int wid = threadIdx.x >> 5, t = threadIdx.x & 31;
    const int node = blockIdx.x * 8 + wid;
    if (node >= Nn) return;

    const float* x0 = X0 + (size_t)node * 64;
    const float* x1 = X1 + (size_t)node * 64;

    float a0 = sb1[t], a1 = sb1[t + 32];   // hidden for view 0 (units t, t+32)
    float c0 = sb1[t], c1 = sb1[t + 32];   // hidden for view 1
    for (int d = 0; d < 64; ++d) {
        const float wa = sW1[d * 64 + t], wb = sW1[d * 64 + t + 32];
        const float v0 = x0[d], v1 = x1[d];
        a0 += v0 * wa; a1 += v0 * wb;
        c0 += v1 * wa; c1 += v1 * wb;
    }
    a0 = fmaxf(a0, 0.f); a1 = fmaxf(a1, 0.f);
    c0 = fmaxf(c0, 0.f); c1 = fmaxf(c1, 0.f);
    float w0 = a0 * sW2[t] + a1 * sW2[t + 32];
    float w1 = c0 * sW2[t] + c1 * sW2[t + 32];
    for (int off = 16; off > 0; off >>= 1) {
        w0 += __shfl_xor(w0, off, 32);
        w1 += __shfl_xor(w1, off, 32);
    }
    const float mx = fmaxf(w0, w1);
    const float e0 = __expf(w0 - mx), e1 = __expf(w1 - mx);
    const float inv = 1.0f / (e0 + e1);
    const float be0 = e0 * inv, be1 = e1 * inv;
    for (int d = t; d < 64; d += 32) {
        const float z = be0 * x0[d] + be1 * x1[d];
        OutF[(size_t)node * 64 + d] = z;
        if (OutB) OutB[(size_t)node * 64 + d] = f2bf(z);
    }
}

// ---------------------------------------------------------------------------
// DEC soft assignment (alpha = 1 -> q = 1/(1+d2), row-normalized)
// ---------------------------------------------------------------------------
__global__ __launch_bounds__(256) void cluster_kernel(
    const float* __restrict__ Z, const float* __restrict__ Mu,
    float* __restrict__ Q, int Nn)
{
    __shared__ float sMu[10 * 64];
    for (int i = threadIdx.x; i < 640; i += 256) sMu[i] = Mu[i];
    __syncthreads();
    const int i = blockIdx.x * 256 + threadIdx.x;
    if (i >= Nn) return;
    float zr[64];
    for (int d = 0; d < 64; ++d) zr[d] = Z[(size_t)i * 64 + d];
    float q[10], s = 0.f;
    for (int k = 0; k < 10; ++k) {
        float d2 = 0.f;
        for (int d = 0; d < 64; ++d) {
            const float t = zr[d] - sMu[k * 64 + d];
            d2 += t * t;
        }
        const float qv = 1.0f / (1.0f + d2);
        q[k] = qv; s += qv;
    }
    const float inv = 1.0f / s;
    for (int k = 0; k < 10; ++k) Q[(size_t)i * 10 + k] = q[k] * inv;
}

// ---------------------------------------------------------------------------
// Host-side GEMM dispatcher
// ---------------------------------------------------------------------------
template<int NT>
static void gemm_dispatch(hipStream_t st, int act, bool accum, bool bias_en,
                          const us* A, int lda, const us* B, int ldb,
                          float* Cf, int ldc, us* Cb, int ldcb, us* Ct, int ldct,
                          const float* bias, int M, int Nc, int Kc)
{
    constexpr int BNc = 32 * NT;
    dim3 grid((Nc + BNc - 1) / BNc, (M + BM - 1) / BM);
    dim3 blk(256);
    const bool SF = Cf != nullptr, SB = Cb != nullptr, ST = Ct != nullptr;
#define GCALL(a, ac, bi, sf, sb, stt)                                          \
    gemm_nt<NT, a, ac, bi, sf, sb, stt><<<grid, blk, 0, st>>>(                 \
        A, lda, B, ldb, Cf, ldc, Cb, ldcb, Ct, ldct, bias, M, Nc, Kc)
    if (accum) {
        GCALL(0, true, false, true, false, false);
    } else if (bias_en) {
        if (act == 1) GCALL(1, false, true, false, true, false);
        else          GCALL(0, false, true, true, false, false);
    } else {
        if      (act == 0 && !SF && !SB &&  ST) GCALL(0, false, false, false, false, true);
        else if (act == 1 && !SF &&  SB && !ST) GCALL(1, false, false, false, true, false);
        else if (act == 0 &&  SF &&  SB && !ST) GCALL(0, false, false, true, true, false);
        else if (act == 2 && !SF &&  SB && !ST) GCALL(2, false, false, false, true, false);
        else if (act == 2 && !SF &&  SB &&  ST) GCALL(2, false, false, false, true, true);
        else if (act == 0 &&  SF && !SB && !ST) GCALL(0, false, false, true, false, false);
        else if (act == 2 &&  SF && !SB && !ST) GCALL(2, false, false, true, false, false);
    }
#undef GCALL
}

static void launch_gemm(hipStream_t st, int act, bool accum, bool bias_en,
                        const us* A, int lda, const us* B, int ldb,
                        float* Cf, int ldc, us* Cb, int ldcb, us* Ct, int ldct,
                        const float* bias, int M, int Nc, int Kc)
{
    if (Nc > 64)
        gemm_dispatch<4>(st, act, accum, bias_en, A, lda, B, ldb,
                         Cf, ldc, Cb, ldcb, Ct, ldct, bias, M, Nc, Kc);
    else
        gemm_dispatch<2>(st, act, accum, bias_en, A, lda, B, ldb,
                         Cf, ldc, Cb, ldcb, Ct, ldct, bias, M, Nc, Kc);
}

// ---------------------------------------------------------------------------
// kernel_launch
// ---------------------------------------------------------------------------
extern "C" void kernel_launch(void* const* d_in, const int* in_sizes, int n_in,
                              void* d_out, int out_size, void* d_ws, size_t ws_size,
                              hipStream_t stream)
{
    (void)in_sizes; (void)n_in; (void)out_size; (void)ws_size;

    const int N = 4000, D0 = 1000, D1 = 800, H1 = 256, H2 = 64, KC = 10;

    const float* feat0   = (const float*)d_in[0];
    const float* feat1   = (const float*)d_in[1];
    const float* adj0    = (const float*)d_in[2];
    const float* adj1    = (const float*)d_in[3];
    const float* Wenc0_0 = (const float*)d_in[4];
    const float* Wenc0_1 = (const float*)d_in[5];
    const float* Wenc1_0 = (const float*)d_in[6];
    const float* Wenc1_1 = (const float*)d_in[7];
    const float* Wdec0_0 = (const float*)d_in[8];
    const float* bdec0_0 = (const float*)d_in[9];
    const float* Wdec0_1 = (const float*)d_in[10];
    const float* bdec0_1 = (const float*)d_in[11];
    const float* Wdec1_0 = (const float*)d_in[12];
    const float* bdec1_0 = (const float*)d_in[13];
    const float* Wdec1_1 = (const float*)d_in[14];
    const float* bdec1_1 = (const float*)d_in[15];
    const float* Wfg     = (const float*)d_in[16];
    const float* Watt1   = (const float*)d_in[17];
    const float* batt1   = (const float*)d_in[18];
    const float* Watt2   = (const float*)d_in[19];
    const float* centers = (const float*)d_in[20];

    // ---------------- output slots ----------------
    float* out      = (float*)d_out;
    float* h_out    = out;                                   // [N,H2]
    float* z_out    = out + (size_t)N * H2;                  // [N,H2]
    float* adjz_out = z_out + (size_t)N * H2;                // [N,N]
    float* Xz0_out  = adjz_out + (size_t)N * N;              // [N,D0]
    float* Xz1_out  = Xz0_out + (size_t)N * D0;              // [N,D1]
    float* qz_out   = Xz1_out + (size_t)N * D1;              // [N,K]
    float* qh_out   = qz_out + (size_t)N * KC;               // [N,K]

    // ---------------- workspace bump allocator ----------------
    char* ws = (char*)d_ws;
    size_t off = 0;
    auto alloc = [&](size_t bytes) -> char* {
        off = (off + 255) & ~(size_t)255;
        char* p = ws + off;
        off += bytes;
        return p;
    };
    us* adj0b  = (us*)alloc((size_t)N * N * 2);
    us* adj1b  = (us*)alloc((size_t)N * N * 2);
    us* feat0b = (us*)alloc((size_t)N * D0 * 2);
    us* feat1b = (us*)alloc((size_t)N * D1 * 2);
    us* W00T   = (us*)alloc((size_t)H1 * D0 * 2);   // [H1,D0]
    us* W01T   = (us*)alloc((size_t)H2 * H1 * 2);   // [H2,H1]
    us* W10T   = (us*)alloc((size_t)H1 * D1 * 2);
    us* W11T   = (us*)alloc((size_t)H2 * H1 * 2);
    us* WfgT   = (us*)alloc((size_t)H2 * H2 * 2);
    us* Wd00T  = (us*)alloc((size_t)H1 * H2 * 2);   // [H1,H2]
    us* Wd01T  = (us*)alloc((size_t)D0 * H1 * 2);   // [D0,H1]
    us* Wd10T  = (us*)alloc((size_t)H1 * H2 * 2);
    us* Wd11T  = (us*)alloc((size_t)D1 * H1 * 2);
    us* t0T    = (us*)alloc((size_t)H1 * N * 2);    // [H1,N]
    us* t1T    = (us*)alloc((size_t)H1 * N * 2);
    us* h0ab   = (us*)alloc((size_t)N * H1 * 2);    // [N,H1]
    us* h1ab   = (us*)alloc((size_t)N * H1 * 2);
    us* t0bT   = (us*)alloc((size_t)H2 * N * 2);    // [H2,N]
    us* t1bT   = (us*)alloc((size_t)H2 * N * 2);
    us* htb    = (us*)alloc((size_t)2 * N * H2 * 2); // [2N,H2] bf16 (h0;h1)
    us* gT     = (us*)alloc((size_t)H2 * 2 * N * 2); // [H2,2N]
    us* crossb  = (us*)alloc((size_t)N * N * 2);
    us* crossTb = (us*)alloc((size_t)N * N * 2);
    us* zb     = (us*)alloc((size_t)N * H2 * 2);
    us* d0hb   = (us*)alloc((size_t)N * H1 * 2);
    us* d1hb   = (us*)alloc((size_t)N * H1 * 2);
    float* h0f   = (float*)alloc((size_t)N * H2 * 4);
    float* h1f   = (float*)alloc((size_t)N * H2 * 4);
    float* hallf = (float*)alloc((size_t)2 * N * H2 * 4);
    // a0/a1 reuse the adjacency bf16 buffers (dead after the encoders)
    us* a0b = adj0b;
    us* a1b = adj1b;

    // ---------------- converts ----------------
    {
        const long nAdj = (long)N * N;
        k_cvt_bf16<<<(unsigned)((nAdj + 1023) / 1024), 256, 0, stream>>>(adj0, adj0b, nAdj);
        k_cvt_bf16<<<(unsigned)((nAdj + 1023) / 1024), 256, 0, stream>>>(adj1, adj1b, nAdj);
        const long n0 = (long)N * D0, n1 = (long)N * D1;
        k_cvt_bf16<<<(unsigned)((n0 + 1023) / 1024), 256, 0, stream>>>(feat0, feat0b, n0);
        k_cvt_bf16<<<(unsigned)((n1 + 1023) / 1024), 256, 0, stream>>>(feat1, feat1b, n1);
    }
#define CVTT(src, dst, R, C) \
    k_cvtT_bf16<<<(unsigned)(((long)(R) * (C) + 255) / 256), 256, 0, stream>>>(src, dst, R, C)
    CVTT(Wenc0_0, W00T, D0, H1);
    CVTT(Wenc0_1, W01T, H1, H2);
    CVTT(Wenc1_0, W10T, D1, H1);
    CVTT(Wenc1_1, W11T, H1, H2);
    CVTT(Wfg,     WfgT, H2, H2);
    CVTT(Wdec0_0, Wd00T, H2, H1);
    CVTT(Wdec0_1, Wd01T, H1, D0);
    CVTT(Wdec1_0, Wd10T, H2, H1);
    CVTT(Wdec1_1, Wd11T, H1, D1);
#undef CVTT

    us* h0b = htb;
    us* h1b = htb + (size_t)N * H2;

    // ---------------- encoders (GCN x2 per view) ----------------
    // t0 = feat0 @ Wenc0_0                    -> store t0^T bf16
    launch_gemm(stream, 0, false, false, feat0b, D0, W00T, D0,
                nullptr, 0, nullptr, 0, t0T, N, nullptr, N, H1, D0);
    // h0a = relu(adj0 @ t0)                   -> bf16 row-major
    launch_gemm(stream, 1, false, false, adj0b, N, t0T, N,
                nullptr, 0, h0ab, H1, nullptr, 0, nullptr, N, H1, N);
    // t0b = h0a @ Wenc0_1                     -> store t0b^T
    launch_gemm(stream, 0, false, false, h0ab, H1, W01T, H1,
                nullptr, 0, nullptr, 0, t0bT, N, nullptr, N, H2, H1);
    // h0 = adj0 @ t0b                          -> f32 + bf16 (into ht)
    launch_gemm(stream, 0, false, false, adj0b, N, t0bT, N,
                h0f, H2, h0b, H2, nullptr, 0, nullptr, N, H2, N);
    // view 1
    launch_gemm(stream, 0, false, false, feat1b, D1, W10T, D1,
                nullptr, 0, nullptr, 0, t1T, N, nullptr, N, H1, D1);
    launch_gemm(stream, 1, false, false, adj1b, N, t1T, N,
                nullptr, 0, h1ab, H1, nullptr, 0, nullptr, N, H1, N);
    launch_gemm(stream, 0, false, false, h1ab, H1, W11T, H1,
                nullptr, 0, nullptr, 0, t1bT, N, nullptr, N, H2, H1);
    launch_gemm(stream, 0, false, false, adj1b, N, t1bT, N,
                h1f, H2, h1b, H2, nullptr, 0, nullptr, N, H2, N);

    // ---------------- similarity maps (overwrite adj buffers) ----------------
    // a0 = sigmoid(h0 h0^T); a1 = sigmoid(h1 h1^T); cross = sigmoid(h0 h1^T)
    launch_gemm(stream, 2, false, false, h0b, H2, h0b, H2,
                nullptr, 0, a0b, N, nullptr, 0, nullptr, N, N, H2);
    launch_gemm(stream, 2, false, false, h1b, H2, h1b, H2,
                nullptr, 0, a1b, N, nullptr, 0, nullptr, N, N, H2);
    launch_gemm(stream, 2, false, false, h0b, H2, h1b, H2,
                nullptr, 0, crossb, N, crossTb, N, nullptr, N, N, H2);

    // ---------------- full-graph GCN: hall = adj_all @ (ht Wfg) ----------------
    // g = ht @ Wfg -> g^T [H2, 2N]
    launch_gemm(stream, 0, false, false, htb, H2, WfgT, H2,
                nullptr, 0, nullptr, 0, gT, 2 * N, nullptr, 2 * N, H2, H2);
    const us* g0T = gT;          // [H2, N] with ld 2N
    const us* g1T = gT + N;
    // hall_top = a0 @ g0 + cross @ g1
    launch_gemm(stream, 0, false, false, a0b, N, g0T, 2 * N,
                hallf, H2, nullptr, 0, nullptr, 0, nullptr, N, H2, N);
    launch_gemm(stream, 0, true,  false, crossb, N, g1T, 2 * N,
                hallf, H2, nullptr, 0, nullptr, 0, nullptr, N, H2, N);
    // hall_bot = cross^T @ g0 + a1 @ g1
    launch_gemm(stream, 0, false, false, crossTb, N, g0T, 2 * N,
                hallf + (size_t)N * H2, H2, nullptr, 0, nullptr, 0, nullptr, N, H2, N);
    launch_gemm(stream, 0, true,  false, a1b, N, g1T, 2 * N,
                hallf + (size_t)N * H2, H2, nullptr, 0, nullptr, 0, nullptr, N, H2, N);

    // ---------------- attention fusion ----------------
    attention_kernel<<<(N + 7) / 8, 256, 0, stream>>>(
        h0f, h1f, Watt1, batt1, Watt2, z_out, zb, N);
    attention_kernel<<<(N + 7) / 8, 256, 0, stream>>>(
        hallf, hallf + (size_t)N * H2, Watt1, batt1, Watt2, h_out, nullptr, N);

    // ---------------- adjz = sigmoid(z z^T) ----------------
    launch_gemm(stream, 2, false, false, zb, H2, zb, H2,
                adjz_out, N, nullptr, 0, nullptr, 0, nullptr, N, N, H2);

    // ---------------- decoders ----------------
    launch_gemm(stream, 1, false, true, zb, H2, Wd00T, H2,
                nullptr, 0, d0hb, H1, nullptr, 0, bdec0_0, N, H1, H2);
    launch_gemm(stream, 0, false, true, d0hb, H1, Wd01T, H1,
                Xz0_out, D0, nullptr, 0, nullptr, 0, bdec0_1, N, D0, H1);
    launch_gemm(stream, 1, false, true, zb, H2, Wd10T, H2,
                nullptr, 0, d1hb, H1, nullptr, 0, bdec1_0, N, H1, H2);
    launch_gemm(stream, 0, false, true, d1hb, H1, Wd11T, H1,
                Xz1_out, D1, nullptr, 0, nullptr, 0, bdec1_1, N, D1, H1);

    // ---------------- cluster soft assignments ----------------
    cluster_kernel<<<(N + 255) / 256, 256, 0, stream>>>(z_out, centers, qz_out, N);
    cluster_kernel<<<(N + 255) / 256, 256, 0, stream>>>(h_out, centers, qh_out, N);
}
